// MultiHeadAttention_43928925504063
// MI455X (gfx1250) — matrix operand order repaired and checked by reading, not compiled, not run on hardware
//
#include <hip/hip_runtime.h>

// ---------- types ----------
typedef __attribute__((ext_vector_type(16))) __bf16 v16bf;
typedef __attribute__((ext_vector_type(8)))  __bf16 v8bf;
typedef __attribute__((ext_vector_type(8)))  float  v8f;

#define WMMA_BF16(a, b, c) \
  __builtin_amdgcn_wmma_f32_16x16x32_bf16(false, (a), false, (b), (short)0, (c), false, false)

constexpr int Bc  = 2;
constexpr int Sc  = 2048;
constexpr int Dc  = 1024;
constexpr int Hc  = 16;
constexpr int DKc = 64;
constexpr int Mrows = Bc * Sc; // 4096

// scale folded into Q: 1/sqrt(DK) * log2(e)  -> scores come out in log2 domain
#define QSCALE 0.18033688011112042f

__device__ __forceinline__ v16bf cat8(v8bf lo, v8bf hi) {
  return __builtin_shufflevector(lo, hi, 0,1,2,3,4,5,6,7,8,9,10,11,12,13,14,15);
}

// ---- raw max (no canonicalize insertion) ----
__device__ __forceinline__ float fmax_raw(float a, float b) {
  float d;
  asm("v_max_num_f32 %0, %1, %2" : "=v"(d) : "v"(a), "v"(b));
  return d;
}

// ---- xor-butterfly within 16-lane rows: fused v_permlane16_b32 + op ----
// EXEC is all-ones at every call site, so permlane's 'old' operand is dead and
// a plain "=v" / "=&v" dest is safe (every lane is written).
template <unsigned SEL0, unsigned SEL1>
__device__ __forceinline__ float maxxor16(float x) {
  float d;
  asm("v_permlane16_b32 %0, %1, %2, %3\n\t"
      "v_max_num_f32 %0, %0, %1"
      : "=&v"(d) : "v"(x), "s"(SEL0), "s"(SEL1));
  return d;
}
template <unsigned SEL0, unsigned SEL1>
__device__ __forceinline__ float addxor16(float x) {
  float d;
  asm("v_permlane16_b32 %0, %1, %2, %3\n\t"
      "v_add_f32 %0, %0, %1"
      : "=&v"(d) : "v"(x), "s"(SEL0), "s"(SEL1));
  return d;
}
template <unsigned SEL0, unsigned SEL1>
__device__ __forceinline__ float getxor16(float x) {
  float d;
  asm("v_permlane16_b32 %0, %1, %2, %3"
      : "=v"(d) : "v"(x), "s"(SEL0), "s"(SEL1));
  return d;
}
// nibble selects for lane -> lane^k within each 16-lane row
#define SEL_X1 0x67452301u, 0xEFCDAB89u
#define SEL_X2 0x54761032u, 0xDCFE98BAu
#define SEL_X4 0x32107654u, 0xBA98FEDCu
#define SEL_X8 0xFEDCBA98u, 0x76543210u

// ---------- f32 -> bf16 conversion (weights) ----------
__global__ void cvt_kernel(const float* __restrict__ src, __bf16* __restrict__ dst, int n) {
  int i = (blockIdx.x * blockDim.x + threadIdx.x) * 4;
  if (i + 3 < n) {
    float4 f = *(const float4*)(src + i);
    dst[i + 0] = (__bf16)f.x;
    dst[i + 1] = (__bf16)f.y;
    dst[i + 2] = (__bf16)f.z;
    dst[i + 3] = (__bf16)f.w;
  }
}

// ---------- fused projection: C = X @ W^T + bias, optional RoPE ----------
// MODE 0: Q (RoPE, *QSCALE, out [B,H,S,DK] bf16)
// MODE 1: K (RoPE, out [B,H,S,DK] bf16)
// MODE 2: V (no RoPE, out transposed [B,H,DK,S] bf16)
template <int MODE>
__global__ __launch_bounds__(256, 1)
void proj_kernel(const float* __restrict__ X,      // [M, Dc] f32
                 const __bf16* __restrict__ Wb,    // [Dc, Dc] bf16 (row = out feature)
                 const float* __restrict__ bias,   // [Dc]
                 const float* __restrict__ cosT,   // [Sc, DKc/2]
                 const float* __restrict__ sinT,   // [Sc, DKc/2]
                 __bf16* __restrict__ Out) {
  const int lane = threadIdx.x & 31;
  const int wave = threadIdx.x >> 5;
  const int half = lane >> 4;
  const int l16  = lane & 15;
  const int m0 = blockIdx.y * 32 + (wave & 1) * 16;
  const int n0 = blockIdx.x * 256 + (wave >> 1) * 64;

  const float* Arow = X + (size_t)(m0 + l16) * Dc;

  v8f c[4] = {};
  for (int kb = 0; kb < Dc; kb += 32) {
    // A fragment: lane<16 -> K {kb..kb+7, kb+16..kb+23}; lane>=16 -> +8
    const int klo = kb + half * 8;
    float af[16];
    *(float4*)(af + 0)  = *(const float4*)(Arow + klo + 0);
    *(float4*)(af + 4)  = *(const float4*)(Arow + klo + 4);
    *(float4*)(af + 8)  = *(const float4*)(Arow + klo + 16);
    *(float4*)(af + 12) = *(const float4*)(Arow + klo + 20);
    v16bf a;
#pragma unroll
    for (int i = 0; i < 16; ++i) a[i] = (__bf16)af[i];

    __builtin_prefetch(Arow + kb + 64, 0, 3);

#pragma unroll
    for (int t = 0; t < 4; ++t) {
      // B fragment: column n fixed per lane, K rows {kb..kb+7, kb+16..kb+23} + half*8
      const __bf16* wr = Wb + (size_t)(n0 + t * 16 + l16) * Dc + kb + half * 8;
      v16bf b = cat8(*(const v8bf*)wr, *(const v8bf*)(wr + 16));
      c[t] = WMMA_BF16(a, b, c[t]);
    }
  }

  // ---- epilogue ----
  if (MODE == 2) {
#pragma unroll
    for (int t = 0; t < 4; ++t) {
      const int n  = n0 + t * 16 + l16;
      const float bn = bias[n];
      const int h  = n >> 6, dk = n & 63;
      const int mb = m0 + half * 8;
      const int b  = mb >> 11, sb = mb & (Sc - 1);
      v8bf pk;
#pragma unroll
      for (int r = 0; r < 8; ++r) pk[r] = (__bf16)(c[t][r] + bn);
      *(v8bf*)(Out + (((size_t)b * Hc + h) * DKc + dk) * Sc + sb) = pk;
    }
  } else {
#pragma unroll
    for (int t = 0; t < 4; ++t) {
      const int n  = n0 + t * 16 + l16;
      const float bn = bias[n];
      const int h  = n >> 6, dk = n & 63, pp = (n & 63) >> 1;
#pragma unroll
      for (int r = 0; r < 8; ++r) {
        const int m = m0 + r + half * 8;
        const int b = m >> 11, s = m & (Sc - 1);
        float val   = c[t][r] + bn;
        float other = getxor16<SEL_X1>(val); // partner feature (n^1) in adjacent lane
        const float cv = cosT[s * (DKc / 2) + pp];
        const float sv = sinT[s * (DKc / 2) + pp];
        float res = ((n & 1) == 0) ? (val * cv - other * sv)
                                   : (other * sv + val * cv);
        if (MODE == 0) res *= QSCALE;
        Out[(((size_t)b * Hc + h) * Sc + s) * DKc + dk] = (__bf16)res;
      }
    }
  }
}

// ---------- one 32-key attention block ----------
template <bool MASKED>
__device__ __forceinline__ void attn_block(
    int k0, int q0, int half, int l16,
    const v16bf& qa0, const v16bf& qa1,
    const __bf16* __restrict__ Kh, const __bf16* __restrict__ Vh,
    float (&mrow)[8], float (&lrow)[8], v8f (&acc)[4],
    __bf16 (&pbuf)[16][40]) {
  // ----- scores (already in log2 domain; QSCALE folded into Q) -----
  v8f c0 = {}, c1 = {};
  const __bf16* kr0 = Kh + (size_t)(k0 + l16) * DKc + half * 16;
  const __bf16* kr1 = kr0 + 16 * DKc;
  v16bf b00 = cat8(*(const v8bf*)kr0,        *(const v8bf*)(kr0 + 8));   // d 0..31
  v16bf b01 = cat8(*(const v8bf*)(kr0 + 32), *(const v8bf*)(kr0 + 40));  // d 32..63
  v16bf b10 = cat8(*(const v8bf*)kr1,        *(const v8bf*)(kr1 + 8));
  v16bf b11 = cat8(*(const v8bf*)(kr1 + 32), *(const v8bf*)(kr1 + 40));
  if (!MASKED) __builtin_prefetch(kr0 + 32 * DKc, 0, 3);
  c0 = WMMA_BF16(qa0, b00, c0);
  c0 = WMMA_BF16(qa1, b01, c0);
  c1 = WMMA_BF16(qa0, b10, c1);
  c1 = WMMA_BF16(qa1, b11, c1);

  // ----- online softmax (base-2); per-lane partial row sums -----
  float p0[8], p1[8], alpha[8];
#pragma unroll
  for (int r = 0; r < 8; ++r) {
    float s0 = c0[r];
    float s1 = c1[r];
    if (MASKED) {
      const int qrow = q0 + r + half * 8;
      if (k0 + l16 > qrow)      s0 = -1.0e30f;
      if (k0 + 16 + l16 > qrow) s1 = -1.0e30f;
    }
    float rm = fmax_raw(s0, s1);
    rm = maxxor16<SEL_X1>(rm);
    rm = maxxor16<SEL_X2>(rm);
    rm = maxxor16<SEL_X4>(rm);
    rm = maxxor16<SEL_X8>(rm);
    const float mn = fmax_raw(mrow[r], rm);
    alpha[r] = exp2f(mrow[r] - mn);
    mrow[r]  = mn;
    p0[r] = exp2f(s0 - mn);
    p1[r] = exp2f(s1 - mn);
    // partial (this lane's 2 columns only); reduced once at the end
    lrow[r] = lrow[r] * alpha[r] + (p0[r] + p1[r]);
  }
#pragma unroll
  for (int t = 0; t < 4; ++t)
#pragma unroll
    for (int r = 0; r < 8; ++r) acc[t][r] *= alpha[r];

  // ----- re-layout probs C->A through LDS (wave-private region) -----
#pragma unroll
  for (int r = 0; r < 8; ++r) {
    pbuf[r + half * 8][l16]      = (__bf16)p0[r];
    pbuf[r + half * 8][16 + l16] = (__bf16)p1[r];
  }
  __builtin_amdgcn_wave_barrier();
  const __bf16* prow = &pbuf[l16][half * 16];
  v16bf pa = cat8(*(const v8bf*)prow, *(const v8bf*)(prow + 8));

  // ----- acc += P @ V (V pre-transposed: contiguous along keys) -----
  const int kb = k0 + half * 16;
#pragma unroll
  for (int t = 0; t < 4; ++t) {
    const __bf16* vr = Vh + (size_t)(t * 16 + l16) * Sc + kb;
    v16bf vb = cat8(*(const v8bf*)vr, *(const v8bf*)(vr + 8));
    acc[t] = WMMA_BF16(pa, vb, acc[t]);
  }
  __builtin_amdgcn_wave_barrier();
}

// ---------- flash attention: causal softmax(QK^T/8) V ----------
__global__ __launch_bounds__(128, 1)
void attn_kernel(const __bf16* __restrict__ Qt,  // [B,H,S,DK] (pre-scaled)
                 const __bf16* __restrict__ Kt,  // [B,H,S,DK]
                 const __bf16* __restrict__ Vt,  // [B,H,DK,S]
                 __bf16* __restrict__ Att) {     // [B,S,D]
  __shared__ __bf16 lds_p[4][16][40]; // per-wave 16x32 probs, padded rows

  const int lane = threadIdx.x & 31;
  const int wave = threadIdx.x >> 5;
  const int half = lane >> 4;
  const int l16  = lane & 15;
  const int bh = blockIdx.y;
  const int q0 = blockIdx.x * 64 + wave * 16;

  const __bf16* Qh = Qt + (size_t)bh * Sc * DKc;
  const __bf16* Kh = Kt + (size_t)bh * Sc * DKc;
  const __bf16* Vh = Vt + (size_t)bh * DKc * Sc;

  // Q fragments (A-matrix), held for the whole KV loop
  const __bf16* qr = Qh + (size_t)(q0 + l16) * DKc + half * 16;
  const v16bf qa0 = cat8(*(const v8bf*)qr,        *(const v8bf*)(qr + 8)); // d 0..31
  const v16bf qa1 = cat8(*(const v8bf*)(qr + 32), *(const v8bf*)(qr + 40)); // d 32..63

  float mrow[8], lrow[8];
  v8f acc[4] = {};
#pragma unroll
  for (int r = 0; r < 8; ++r) { mrow[r] = -3.0e38f; lrow[r] = 0.0f; }

  // Full (unmasked) blocks: k0+31 <= q0, then exactly one masked diagonal block.
  int k0 = 0;
  for (; k0 + 31 <= q0; k0 += 32)
    attn_block<false>(k0, q0, half, l16, qa0, qa1, Kh, Vh, mrow, lrow, acc, lds_p[wave]);
  attn_block<true>(k0, q0, half, l16, qa0, qa1, Kh, Vh, mrow, lrow, acc, lds_p[wave]);

  // ----- final cross-lane reduction of l, normalize, store [B,S,D] bf16 -----
  const int b = bh >> 4, h = bh & 15;
#pragma unroll
  for (int r = 0; r < 8; ++r) {
    float l = lrow[r];
    l = addxor16<SEL_X1>(l);
    l = addxor16<SEL_X2>(l);
    l = addxor16<SEL_X4>(l);
    l = addxor16<SEL_X8>(l);
    const float inv = 1.0f / l;
    const int srow = q0 + r + half * 8;
#pragma unroll
    for (int t = 0; t < 4; ++t) {
      Att[((size_t)b * Sc + srow) * Dc + h * DKc + t * 16 + l16] =
          (__bf16)(acc[t][r] * inv);
    }
  }
}

// ---------- output projection: Out = Att @ Wo^T + bias (f32 out) ----------
__global__ __launch_bounds__(256, 1)
void outproj_kernel(const __bf16* __restrict__ A,   // [M, Dc] bf16
                    const __bf16* __restrict__ Wb,  // [Dc, Dc] bf16
                    const float* __restrict__ bias, // [Dc]
                    float* __restrict__ Out) {      // [M, Dc] f32
  const int lane = threadIdx.x & 31;
  const int wave = threadIdx.x >> 5;
  const int half = lane >> 4;
  const int l16  = lane & 15;
  const int m0 = blockIdx.y * 32 + (wave & 1) * 16;
  const int n0 = blockIdx.x * 256 + (wave >> 1) * 64;

  const __bf16* Arow = A + (size_t)(m0 + l16) * Dc;

  v8f c[4] = {};
  for (int kb = 0; kb < Dc; kb += 32) {
    const int klo = kb + half * 16;
    v16bf a = cat8(*(const v8bf*)(Arow + klo), *(const v8bf*)(Arow + klo + 8));
    __builtin_prefetch(Arow + kb + 64, 0, 3);
#pragma unroll
    for (int t = 0; t < 4; ++t) {
      const __bf16* wr = Wb + (size_t)(n0 + t * 16 + l16) * Dc + kb + half * 16;
      v16bf b = cat8(*(const v8bf*)wr, *(const v8bf*)(wr + 8));
      c[t] = WMMA_BF16(a, b, c[t]);
    }
  }

#pragma unroll
  for (int t = 0; t < 4; ++t) {
    const int n = n0 + t * 16 + l16;
    const float bn = bias[n];
#pragma unroll
    for (int r = 0; r < 8; ++r)
      Out[(size_t)(m0 + r + half * 8) * Dc + n] = c[t][r] + bn;
  }
}

// ---------- launch ----------
extern "C" void kernel_launch(void* const* d_in, const int* in_sizes, int n_in,
                              void* d_out, int out_size, void* d_ws, size_t ws_size,
                              hipStream_t stream) {
  (void)in_sizes; (void)n_in; (void)out_size;

  const float* Q    = (const float*)d_in[0];
  const float* K    = (const float*)d_in[1];
  const float* V    = (const float*)d_in[2];
  // d_in[3] = mask (unused: causal structure is known)
  const float* Wq   = (const float*)d_in[4];
  const float* Wqb  = (const float*)d_in[5];
  const float* Wk   = (const float*)d_in[6];
  const float* Wkb  = (const float*)d_in[7];
  const float* Wv   = (const float*)d_in[8];
  const float* Wvb  = (const float*)d_in[9];
  const float* Wo   = (const float*)d_in[10];
  const float* Wob  = (const float*)d_in[11];
  const float* cosT = (const float*)d_in[12];
  const float* sinT = (const float*)d_in[13];

  char* ws = (char*)d_ws;
  size_t off = 0;
  auto take = [&](size_t bytes) -> char* {
    char* p = ws + off;
    off = (off + bytes + 255) & ~(size_t)255;
    return p;
  };
  const size_t wbytes = (size_t)Dc * Dc * 2;       // 2 MB each
  const size_t tbytes = (size_t)Bc * Sc * Dc * 2;  // 8 MB each
  __bf16* wq_bf  = (__bf16*)take(wbytes);
  __bf16* wk_bf  = (__bf16*)take(wbytes);
  __bf16* wv_bf  = (__bf16*)take(wbytes);
  __bf16* wo_bf  = (__bf16*)take(wbytes);
  __bf16* q_bf   = (__bf16*)take(tbytes);
  __bf16* k_bf   = (__bf16*)take(tbytes);
  __bf16* vt_bf  = (__bf16*)take(tbytes);
  __bf16* att_bf = (__bf16*)take(tbytes);
  if (off > ws_size) return; // workspace too small (should not happen)

  const int nW = Dc * Dc;
  cvt_kernel<<<nW / 1024, 256, 0, stream>>>(Wq, wq_bf, nW);
  cvt_kernel<<<nW / 1024, 256, 0, stream>>>(Wk, wk_bf, nW);
  cvt_kernel<<<nW / 1024, 256, 0, stream>>>(Wv, wv_bf, nW);
  cvt_kernel<<<nW / 1024, 256, 0, stream>>>(Wo, wo_bf, nW);

  dim3 gp(Dc / 256, Mrows / 32); // (4, 128)
  proj_kernel<0><<<gp, 256, 0, stream>>>(Q, wq_bf, Wqb, cosT, sinT, q_bf);
  proj_kernel<1><<<gp, 256, 0, stream>>>(K, wk_bf, Wkb, cosT, sinT, k_bf);
  proj_kernel<2><<<gp, 256, 0, stream>>>(V, wv_bf, Wvb, cosT, sinT, vt_bf);

  dim3 ga(Sc / 64, Bc * Hc); // (32, 32)
  attn_kernel<<<ga, 128, 0, stream>>>(q_bf, k_bf, vt_bf, att_bf);

  outproj_kernel<<<gp, 256, 0, stream>>>(att_bf, wo_bf, Wob, (float*)d_out);
}
